// QuaternionAttention_6983616823301
// MI455X (gfx1250) — matrix operand order repaired and checked by reading, not compiled, not run on hardware
//
#include <hip/hip_runtime.h>

typedef __attribute__((ext_vector_type(16))) _Float16 v16h;
typedef __attribute__((ext_vector_type(8)))  float    v8f;

#define B_    4
#define L_    2048
#define S_    2048
#define H_    8
#define E_    64
#define M_    4
#define LTILE 64
#define STILE 64
#define NW    4
#define NT    128

union FragH { uint4 q[2]; v16h v; };

__device__ __forceinline__ v16h ld_frag_pair(const _Float16* p0, const _Float16* p1) {
  FragH f;
  f.q[0] = *reinterpret_cast<const uint4*>(p0);
  f.q[1] = *reinterpret_cast<const uint4*>(p1);
  return f.v;
}
__device__ __forceinline__ v16h ld_frag_contig(const _Float16* p) {
  FragH f;
  f.q[0] = reinterpret_cast<const uint4*>(p)[0];
  f.q[1] = reinterpret_cast<const uint4*>(p)[1];
  return f.v;
}

__global__ __launch_bounds__(NT)
void quat_attn_flash_kernel(const float* __restrict__ Q, const float* __restrict__ K,
                            const float* __restrict__ V,
                            const float* __restrict__ QO, const float* __restrict__ QT,
                            const float* __restrict__ KO, const float* __restrict__ KT,
                            float* __restrict__ Out)
{
  __shared__ _Float16 Qr[LTILE * 256];      // rotated Q tile  [row][k]  32 KB
  __shared__ _Float16 Kr[STILE * 256];      // rotated K tile  [col][k]  32 KB
  __shared__ _Float16 Vt[E_ * STILE];       // V transposed    [e][s]     8 KB
  __shared__ _Float16 Pl[NW * 16 * STILE];  // per-wave P      [row][s]   8 KB

  const int tid  = threadIdx.x;
  const int lane = tid & 31;
  const int wave = tid >> 5;
  const int g    = lane >> 4;   // half-wave group -> K-range of fragments
  const int lm   = lane & 15;

  const int nLt   = L_ / LTILE;             // 32
  const int lt    = blockIdx.x % nLt;
  const int bh    = blockIdx.x / nLt;
  const int b     = bh / H_;
  const int h     = bh % H_;
  const int qBase = lt * LTILE;

  const float SC = 1.0f / 32.0f;            // 1/(M*sqrt(E))

  // ---------- Phase 1: rotated Q tile (64 x 256) into LDS, pre-scaled ----------
  for (int item = tid; item < LTILE * M_; item += NT) {
    const int r = item >> 2;
    const int m = item & 3;
    const int l = qBase + r;
    const float pos = (float)l * (1.0f / (float)L_);
    const int ob = ((b * L_ + l) * H_ + h) * M_ + m;
    const float ang = QO[ob] * pos + QT[ob];
    const float c = __cosf(ang);
    const float s = __sinf(ang);
    const float* qrow = Q + (((size_t)b * L_ + l) * H_ + h) * E_;
    union { _Float16 hx[64]; uint4 q4[8]; } t;
#pragma unroll
    for (int e = 0; e < 16; ++e) {
      const float a0 = qrow[e], a1 = qrow[16 + e], a2 = qrow[32 + e], a3 = qrow[48 + e];
      t.hx[e]      = (_Float16)((a0 * c - a1 * s) * SC);
      t.hx[16 + e] = (_Float16)((a1 * c + a0 * s) * SC);
      t.hx[32 + e] = (_Float16)((a2 * c + a3 * s) * SC);
      t.hx[48 + e] = (_Float16)((a3 * c - a2 * s) * SC);
    }
    uint4* dst = reinterpret_cast<uint4*>(&Qr[r * 256 + m * 64]);
#pragma unroll
    for (int i = 0; i < 8; ++i) dst[i] = t.q4[i];
  }
  __syncthreads();

  // ---------- this wave's Q A-fragments: 8 x (16x32) in registers ----------
  const int rowBase = wave * 16;
  v16h qa[8];
#pragma unroll
  for (int kk = 0; kk < 8; ++kk) {
    const _Float16* base = &Qr[(rowBase + lm) * 256 + kk * 32 + g * 8];
    qa[kk] = ld_frag_pair(base, base + 16);   // K = g*8..+7 and 16+g*8..+7
  }

  float rmax[8], rsum[8];
  v8f o[4];
  const v8f vzero = {};
#pragma unroll
  for (int r = 0; r < 8; ++r) { rmax[r] = -3.0e30f; rsum[r] = 0.0f; }
#pragma unroll
  for (int ne = 0; ne < 4; ++ne) o[ne] = vzero;

  _Float16* plw = &Pl[wave * 16 * STILE];

  for (int sBase = 0; sBase < S_; sBase += STILE) {
    __syncthreads();  // previous K/V tile fully consumed

    // prefetch next tile (global_prefetch_b8)
    if (sBase + STILE < S_) {
      const int pr = tid >> 1;
      const size_t kidx = (((size_t)b * S_ + (sBase + STILE + pr)) * H_ + h) * E_ + (size_t)(tid & 1) * 32;
      __builtin_prefetch(K + kidx, 0, 1);
      __builtin_prefetch(V + kidx, 0, 1);
    }

    // ---------- rotated K tile (64 x 256): 256 (row,m) items ----------
    for (int item = tid; item < STILE * M_; item += NT) {
      const int r = item >> 2;
      const int m = item & 3;
      const int sg = sBase + r;
      const float pos = (float)sg * (1.0f / (float)S_);
      const int ob = ((b * S_ + sg) * H_ + h) * M_ + m;
      const float ang = KO[ob] * pos + KT[ob];
      const float c = __cosf(ang);
      const float s = __sinf(ang);
      const float* krow = K + (((size_t)b * S_ + sg) * H_ + h) * E_;
      union { _Float16 hx[64]; uint4 q4[8]; } t;
#pragma unroll
      for (int e = 0; e < 16; ++e) {
        const float a0 = krow[e], a1 = krow[16 + e], a2 = krow[32 + e], a3 = krow[48 + e];
        t.hx[e]      = (_Float16)(a0 * c - a2 * s);
        t.hx[16 + e] = (_Float16)(a1 * c - a3 * s);
        t.hx[32 + e] = (_Float16)(a2 * c + a0 * s);
        t.hx[48 + e] = (_Float16)(a3 * c + a1 * s);
      }
      uint4* dst = reinterpret_cast<uint4*>(&Kr[r * 256 + m * 64]);
#pragma unroll
      for (int i = 0; i < 8; ++i) dst[i] = t.q4[i];
    }

    // ---------- V tile transposed into LDS: Vt[e][s], 64x64 ----------
    {
      const int e  = tid & 63;
      const int s0 = (tid >> 6) * 32;
#pragma unroll
      for (int j = 0; j < 32; ++j) {
        const int sv = s0 + j;
        Vt[e * STILE + sv] =
            (_Float16)V[(((size_t)b * S_ + sBase + sv) * H_ + h) * E_ + e];
      }
    }
    __syncthreads();

    // ---------- scores 16x64: 4 N-slices x 8 K-steps = 32 WMMAs ----------
    v8f sc[4];
#pragma unroll
    for (int ns = 0; ns < 4; ++ns) {
      v8f acc = vzero;
#pragma unroll
      for (int kk = 0; kk < 8; ++kk) {
        const _Float16* kb = &Kr[(ns * 16 + lm) * 256 + kk * 32 + g * 8];
        const v16h bf = ld_frag_pair(kb, kb + 16);
        acc = __builtin_amdgcn_wmma_f32_16x16x32_f16(false, qa[kk], false, bf,
                                                     (short)0, acc, false, false);
      }
      sc[ns] = acc;
    }

    // ---------- online softmax (rows live in half-wave lane groups) ----------
    float mnew[8], corr[8];
#pragma unroll
    for (int r = 0; r < 8; ++r) {
      float mx = fmaxf(fmaxf(sc[0][r], sc[1][r]), fmaxf(sc[2][r], sc[3][r]));
#pragma unroll
      for (int msk = 1; msk <= 8; msk <<= 1)
        mx = fmaxf(mx, __shfl_xor(mx, msk, 32));
      mnew[r] = fmaxf(rmax[r], mx);
      corr[r] = __expf(rmax[r] - mnew[r]);
      rmax[r] = mnew[r];
    }
#pragma unroll
    for (int r = 0; r < 8; ++r) {
      const float p0 = __expf(sc[0][r] - mnew[r]);
      const float p1 = __expf(sc[1][r] - mnew[r]);
      const float p2 = __expf(sc[2][r] - mnew[r]);
      const float p3 = __expf(sc[3][r] - mnew[r]);
      _Float16* prow = &plw[(r + 8 * g) * STILE + lm];
      prow[0]  = (_Float16)p0;
      prow[16] = (_Float16)p1;
      prow[32] = (_Float16)p2;
      prow[48] = (_Float16)p3;
      float ps = (p0 + p1) + (p2 + p3);
#pragma unroll
      for (int msk = 1; msk <= 8; msk <<= 1)
        ps += __shfl_xor(ps, msk, 32);
      rsum[r] = rsum[r] * corr[r] + ps;
    }
#pragma unroll
    for (int ne = 0; ne < 4; ++ne)
#pragma unroll
      for (int r = 0; r < 8; ++r)
        o[ne][r] *= corr[r];

    // ---------- P @ V : reshape P via LDS, K=64 -> 2 frags, 8 WMMAs ----------
    const _Float16* pb0 = &plw[lm * STILE + g * 8];
    const _Float16* pb1 = &plw[lm * STILE + 32 + g * 8];
    const v16h pf0 = ld_frag_pair(pb0, pb0 + 16);
    const v16h pf1 = ld_frag_pair(pb1, pb1 + 16);
#pragma unroll
    for (int ne = 0; ne < 4; ++ne) {
      const _Float16* vb0 = &Vt[(ne * 16 + lm) * STILE + g * 8];
      const _Float16* vb1 = &Vt[(ne * 16 + lm) * STILE + 32 + g * 8];
      const v16h vf0 = ld_frag_pair(vb0, vb0 + 16);
      const v16h vf1 = ld_frag_pair(vb1, vb1 + 16);
      o[ne] = __builtin_amdgcn_wmma_f32_16x16x32_f16(false, pf0, false, vf0,
                                                     (short)0, o[ne], false, false);
      o[ne] = __builtin_amdgcn_wmma_f32_16x16x32_f16(false, pf1, false, vf1,
                                                     (short)0, o[ne], false, false);
    }
  }

  // ---------- epilogue: normalize and store ----------
#pragma unroll
  for (int r = 0; r < 8; ++r) {
    const float inv = 1.0f / rsum[r];
    const int row = qBase + rowBase + r + 8 * g;
    float* op = Out + (((size_t)b * L_ + row) * H_ + h) * E_ + lm;
#pragma unroll
    for (int ne = 0; ne < 4; ++ne)
      op[ne * 16] = o[ne][r] * inv;
  }
}

extern "C" void kernel_launch(void* const* d_in, const int* in_sizes, int n_in,
                              void* d_out, int out_size, void* d_ws, size_t ws_size,
                              hipStream_t stream) {
  const float* Q  = (const float*)d_in[0];
  const float* K  = (const float*)d_in[1];
  const float* V  = (const float*)d_in[2];
  const float* QO = (const float*)d_in[3];
  const float* QT = (const float*)d_in[4];
  const float* KO = (const float*)d_in[5];
  const float* KT = (const float*)d_in[6];
  float* Out = (float*)d_out;

  dim3 grid(B_ * H_ * (L_ / LTILE));   // 1024 workgroups
  dim3 block(NT);                      // 4 waves (wave32)
  quat_attn_flash_kernel<<<grid, block, 0, stream>>>(Q, K, V, QO, QT, KO, KT, Out);
}
